// uw2V_for_restore_88098369175596
// MI455X (gfx1250) — matrix operand order repaired and checked by reading, not compiled, not run on hardware
//
#include <hip/hip_runtime.h>

typedef __attribute__((ext_vector_type(16))) _Float16 v16h;
typedef __attribute__((ext_vector_type(8)))  float    v8f;

#define NTT 64
#define SIG 0.1f
#define H_STRIDE 65536
#define PRED_STRIDE 2560
#define NU 2048

// ---- WMMA operand loaders (wave32, v_wmma_f32_16x16x32_f16) ----
// A-matrix 16x32 f16: lane<16 holds K 0..15 of row (lane&15);
// lane>=16 holds K 16..31.
__device__ inline v16h loadA_pat(const _Float16* base, int t0, int j0) {
  int lane = threadIdx.x & 31;
  const _Float16* p = base + (t0 + (lane & 15)) * 256 + j0 + ((lane >> 4) << 4);
  v16h v;
#pragma unroll
  for (int i = 0; i < 16; ++i) v[i] = p[i];
  return v;
}

// B-matrix 32x16 f16: lane&15 = column, lane>>4 selects K 0..15 vs 16..31.
__device__ inline v16h loadB_row(const _Float16* base, int s0, int j0) {
  int lane = threadIdx.x & 31;
  const _Float16* p = base + (s0 + (lane & 15)) * 256 + j0 + ((lane >> 4) << 4);
  v16h v;
#pragma unroll
  for (int i = 0; i < 16; ++i) v[i] = p[i];
  return v;
}

__global__ __launch_bounds__(256) void beamform_fused_kernel(
    const float* __restrict__ chan, const float* __restrict__ pred,
    float* __restrict__ out) {
  // ---- LDS (~148 KB) ----
  __shared__ _Float16 Mre[NTT * 256];        // HHU real, f16   (32 KB)
  __shared__ _Float16 Mim[NTT * 256];        // HHU imag, f16   (32 KB)
  __shared__ _Float16 Nbuf[NTT * 256];       // Nre, then Nim   (32 KB)
  __shared__ float Bre[NTT * 65];            // temp_B real     (16.6 KB)
  __shared__ float Bim[NTT * 65];
  __shared__ float Xre[NTT * 16];            // HUW -> solution (4 KB)
  __shared__ float Xim[NTT * 16];
  __shared__ __align__(16) float Ulds[NU];   // raw U floats    (8 KB)
  __shared__ float Wlds[512];                // W (real)        (2 KB)
  __shared__ float trS[2];
  __shared__ float nrm2s;

  const int tid = threadIdx.x;
  const int b = blockIdx.x;
  const float* Hb = chan + (size_t)b * H_STRIDE;
  const float* Pb = pred + (size_t)b * PRED_STRIDE;

  // ---- init ----
  for (int i = tid; i < NU; i += 256) Ulds[i] = Pb[i];
  for (int i = tid; i < 512; i += 256) Wlds[i] = Pb[NU + i];
  for (int i = tid; i < NTT * 16; i += 256) { Xre[i] = 0.f; Xim[i] = 0.f; }
  if (tid == 0) { trS[0] = 0.f; trS[1] = 0.f; nrm2s = 0.f; }
  __syncthreads();

  // ---- tr_UWU = sum_{r,d,e,f,k} U[r,d] W[d,e] conj(U[r,e]) ----
  for (int i = tid; i < 512; i += 256) {
    int r = i >> 7, f = (i >> 3) & 15, k = i & 7;
    const float* u0 = Ulds + (((r * 2 + 0) * 16 + f) << 4);
    const float* u1 = Ulds + (((r * 2 + 1) * 16 + f) << 4);
    float u0r = u0[k], u0i = u0[k + 8], u1r = u1[k], u1i = u1[k + 8];
    int fk = f * 8 + k;
    float W00 = Wlds[fk], W01 = Wlds[128 + fk], W10 = Wlds[256 + fk], W11 = Wlds[384 + fk];
    float tre = W00 * (u0r * u0r + u0i * u0i) + W11 * (u1r * u1r + u1i * u1i);
    float cr = u0r * u1r + u0i * u1i;
    float ci = u0i * u1r - u0r * u1i;
    tre += (W01 + W10) * cr;
    float tim = (W01 - W10) * ci;
    atomicAdd(&trS[0], tre);
    atomicAdd(&trS[1], tim);
  }

  // ---- stage A: HHU -> M (f16), Nre -> Nbuf (f16), HUW -> X (LDS atomics) ----
  for (int pair = tid; pair < NTT * 16; pair += 256) {
    int t = pair >> 4, f = pair & 15;
    float ar[2][8], ai[2][8];
#pragma unroll
    for (int d = 0; d < 2; ++d)
#pragma unroll
      for (int k = 0; k < 8; ++k) { ar[d][k] = 0.f; ai[d][k] = 0.f; }
#pragma unroll
    for (int r = 0; r < 4; ++r) {
      const float* Hp = Hb + (((t * 4 + r) * 16 + f) << 4);
      __builtin_prefetch(Hp + 1024, 0, 3);
#pragma unroll
      for (int d = 0; d < 2; ++d) {
        const float* Up = Ulds + (((r * 2 + d) * 16 + f) << 4);
#pragma unroll
        for (int k = 0; k < 8; ++k) {
          float hr = Hp[k], hi = Hp[k + 8];
          float ur = Up[k], ui = Up[k + 8];
          ar[d][k] += hr * ur + hi * ui;   // conj(H)*U
          ai[d][k] += hr * ui - hi * ur;
        }
      }
    }
    int mb = t * 256 + f * 8;
#pragma unroll
    for (int k = 0; k < 8; ++k) {
#pragma unroll
      for (int d = 0; d < 2; ++d) {
        Mre[mb + d * 128 + k] = (_Float16)ar[d][k];
        Mim[mb + d * 128 + k] = (_Float16)ai[d][k];
      }
#pragma unroll
      for (int e = 0; e < 2; ++e) {
        float w0 = Wlds[e * 128 + f * 8 + k];
        float w1 = Wlds[(2 + e) * 128 + f * 8 + k];
        float nre = ar[0][k] * w0 + ar[1][k] * w1;
        float nim = ai[0][k] * w0 + ai[1][k] * w1;
        Nbuf[mb + e * 128 + k] = (_Float16)nre;      // Nre
        atomicAdd(&Xre[t * 16 + e * 8 + k], nre);
        atomicAdd(&Xim[t * 16 + e * 8 + k], nim);
      }
    }
  }
  __syncthreads();

  // ---- stage C: quad = N @ M^H (64x64x256 complex) via WMMA, two passes ----
  // Re = Nre.Mre^T + Nim.Mim^T ; Im = Nim.Mre^T - Nre.Mim^T
  const int wave = tid >> 5;
  const int lane = tid & 31;
  const int ti = wave >> 1;            // both tiles of a wave share the row block
  const int si0 = (wave & 1) * 2;
  v8f accRe[2] = {{0, 0, 0, 0, 0, 0, 0, 0}, {0, 0, 0, 0, 0, 0, 0, 0}};
  v8f accP[2]  = {{0, 0, 0, 0, 0, 0, 0, 0}, {0, 0, 0, 0, 0, 0, 0, 0}};
  v8f accQ[2]  = {{0, 0, 0, 0, 0, 0, 0, 0}, {0, 0, 0, 0, 0, 0, 0, 0}};

  // pass 1: Nbuf holds Nre
#pragma unroll
  for (int j0 = 0; j0 < 256; j0 += 32) {
    v16h a = loadA_pat(Nbuf, ti * 16, j0);
#pragma unroll
    for (int tt = 0; tt < 2; ++tt) {
      v16h bre = loadB_row(Mre, (si0 + tt) * 16, j0);
      v16h bim = loadB_row(Mim, (si0 + tt) * 16, j0);
      accRe[tt] = __builtin_amdgcn_wmma_f32_16x16x32_f16(false, a, false, bre, (short)0, accRe[tt], false, false);
      accQ[tt]  = __builtin_amdgcn_wmma_f32_16x16x32_f16(false, a, false, bim, (short)0, accQ[tt],  false, false);
    }
  }
  __syncthreads();

  // refill Nbuf with Nim = Mim(d=0)*W[0,e] + Mim(d=1)*W[1,e]
  for (int i = tid; i < NTT * 256; i += 256) {
    int t = i >> 8, j = i & 255, e = j >> 7, fk = j & 127;
    float w0 = Wlds[e * 128 + fk];
    float w1 = Wlds[(2 + e) * 128 + fk];
    Nbuf[i] = (_Float16)((float)Mim[t * 256 + fk] * w0 + (float)Mim[t * 256 + 128 + fk] * w1);
  }
  __syncthreads();

  // pass 2: Nbuf holds Nim
#pragma unroll
  for (int j0 = 0; j0 < 256; j0 += 32) {
    v16h a = loadA_pat(Nbuf, ti * 16, j0);
#pragma unroll
    for (int tt = 0; tt < 2; ++tt) {
      v16h bre = loadB_row(Mre, (si0 + tt) * 16, j0);
      v16h bim = loadB_row(Mim, (si0 + tt) * 16, j0);
      accRe[tt] = __builtin_amdgcn_wmma_f32_16x16x32_f16(false, a, false, bim, (short)0, accRe[tt], false, false);
      accP[tt]  = __builtin_amdgcn_wmma_f32_16x16x32_f16(false, a, false, bre, (short)0, accP[tt],  false, false);
    }
  }

  // write temp_B = quad + SIG*tr*I
#pragma unroll
  for (int tt = 0; tt < 2; ++tt) {
    int col = (si0 + tt) * 16 + (lane & 15);
#pragma unroll
    for (int g = 0; g < 8; ++g) {
      int row = ti * 16 + ((lane >> 4) << 3) + g;
      float re = accRe[tt][g];
      float im = accP[tt][g] - accQ[tt][g];
      if (row == col) { re += SIG * trS[0]; im += SIG * trS[1]; }
      Bre[row * 65 + col] = re;
      Bim[row * 65 + col] = im;
    }
  }
  __syncthreads();

  // ---- stage D: Gauss-Jordan on augmented [B | X] (complex, no pivoting) ----
  for (int p = 0; p < NTT; ++p) {
    // scale pivot row (cols > p plus 16 RHS cols) by 1/B[p][p]
    {
      float brr = Bre[p * 65 + p], bii = Bim[p * 65 + p];
      float inv = 1.0f / (brr * brr + bii * bii);
      float dr = brr * inv, di = -bii * inv;
      int ncB = NTT - 1 - p;
      for (int c = tid; c < ncB + 16; c += 256) {
        if (c < ncB) {
          int cc = p + 1 + c;
          float ur = Bre[p * 65 + cc], ui = Bim[p * 65 + cc];
          Bre[p * 65 + cc] = ur * dr - ui * di;
          Bim[p * 65 + cc] = ur * di + ui * dr;
        } else {
          int cc = c - ncB;
          float ur = Xre[p * 16 + cc], ui = Xim[p * 16 + cc];
          Xre[p * 16 + cc] = ur * dr - ui * di;
          Xim[p * 16 + cc] = ur * di + ui * dr;
        }
      }
    }
    __syncthreads();
    // eliminate all rows r != p
    {
      int ncB = NTT - 1 - p;
      int ncols = ncB + 16;
      int total = (NTT - 1) * ncols;
      for (int idx = tid; idx < total; idx += 256) {
        int r = idx / ncols;
        if (r >= p) r++;
        int c = idx % ncols;
        float lr = Bre[r * 65 + p], li = Bim[r * 65 + p];
        if (c < ncB) {
          int cc = p + 1 + c;
          float ur = Bre[p * 65 + cc], ui = Bim[p * 65 + cc];
          Bre[r * 65 + cc] -= lr * ur - li * ui;
          Bim[r * 65 + cc] -= lr * ui + li * ur;
        } else {
          int cc = c - ncB;
          float ur = Xre[p * 16 + cc], ui = Xim[p * 16 + cc];
          Xre[r * 16 + cc] -= lr * ur - li * ui;
          Xim[r * 16 + cc] -= lr * ui + li * ur;
        }
      }
    }
    __syncthreads();
  }

  // ---- normalize (sqrt(P)/||V0||, P=1) and write out ----
  float local = 0.f;
  for (int i = tid; i < NTT * 16; i += 256)
    local += Xre[i] * Xre[i] + Xim[i] * Xim[i];
  atomicAdd(&nrm2s, local);
  __syncthreads();
  float s = rsqrtf(nrm2s);
  for (int i = tid; i < NTT * 16; i += 256) {
    out[((size_t)b * 1024 + i) * 2 + 0] = Xre[i] * s;
    out[((size_t)b * 1024 + i) * 2 + 1] = Xim[i] * s;
  }
}

extern "C" void kernel_launch(void* const* d_in, const int* in_sizes, int n_in,
                              void* d_out, int out_size, void* d_ws, size_t ws_size,
                              hipStream_t stream) {
  const float* chan = (const float*)d_in[0];
  const float* pred = (const float*)d_in[1];
  float* out = (float*)d_out;
  (void)in_sizes; (void)n_in; (void)out_size; (void)d_ws; (void)ws_size;
  beamform_fused_kernel<<<dim3(256), dim3(256), 0, stream>>>(chan, pred, out);
}
